// CausalSelfAttention_14104672600135
// MI455X (gfx1250) — matrix operand order repaired and checked by reading, not compiled, not run on hardware
//
#include <hip/hip_runtime.h>
#include <hip/hip_bf16.h>
#include <math.h>

// Problem constants: B=4, T=2048, C=1024, H=16, D=64
#define TB 2048
#define CB 1024
#define DB 64

typedef __attribute__((ext_vector_type(8)))  __bf16 v8bf;
typedef __attribute__((ext_vector_type(16))) __bf16 v16bf;
typedef __attribute__((ext_vector_type(8)))  float  v8f;
typedef __attribute__((ext_vector_type(4)))  unsigned int u32x4;
typedef __attribute__((ext_vector_type(8)))  int i32x8;
typedef __attribute__((ext_vector_type(4)))  int i32x4;

__device__ __forceinline__ __bf16 f2bf(float f) {
    unsigned u = __builtin_bit_cast(unsigned, f);
    unsigned r = (u + 0x7FFFu + ((u >> 16) & 1u)) >> 16;
    return __builtin_bit_cast(__bf16, (unsigned short)r);
}

__device__ __forceinline__ v16bf cat16(v8bf lo, v8bf hi) {
    v16bf r;
#pragma unroll
    for (int j = 0; j < 8; ++j) { r[j] = lo[j]; r[j + 8] = hi[j]; }
    return r;
}

__device__ __forceinline__ v8f wmma_bf16(v16bf a, v16bf b, v8f c) {
    return __builtin_amdgcn_wmma_f32_16x16x32_bf16(false, a, false, b,
                                                   (short)0, c, false, false);
}

// A fragment (16x32 bf16): lane m=lane&15 holds row m; K halves {h*8..h*8+7, 16+h*8..}
__device__ __forceinline__ v16bf load_afrag(const __bf16* src, int pitch, int lane) {
    int m = lane & 15, half = lane >> 4;
    const __bf16* p = src + (size_t)m * pitch + half * 8;
    v8bf lo = *reinterpret_cast<const v8bf*>(p);
    v8bf hi = *reinterpret_cast<const v8bf*>(p + 16);
    return cat16(lo, hi);
}

// B fragment (32x16 bf16, stored as [N][K] K-contig): lane n=lane&15 col, K halves {kh*8..kh*8+7, 16+kh*8..}
__device__ __forceinline__ v16bf load_bfrag(const __bf16* src, int pitch, int lane) {
    int n = lane & 15, kh = lane >> 4;
    const __bf16* p = src + (size_t)n * pitch + kh * 8;
    v8bf lo = *reinterpret_cast<const v8bf*>(p);
    v8bf hi = *reinterpret_cast<const v8bf*>(p + 16);
    return cat16(lo, hi);
}

// TDM: stage a 128-row x 32-col bf16 tile (row pitch `gPitch` elems in global) into LDS,
// HW-padding each 64B row with 16B so the LDS pitch is 40 bf16 (matches fragment loaders).
// D# per CDNA5 ISA. One issue per wave; tracked by TENSORcnt.
__device__ __forceinline__ void tdm_load_tile_bf16(const __bf16* gsrc, unsigned ldsByteAddr) {
    unsigned long long ga = (unsigned long long)(size_t)gsrc;
    u32x4 g0;
    g0[0] = 1u;                                        // count=1, user mode
    g0[1] = ldsByteAddr;                               // lds_addr [63:32]
    g0[2] = (unsigned)(ga & 0xFFFFFFFFu);              // global_addr low
    g0[3] = (unsigned)((ga >> 32) & 0x01FFFFFFu) | (2u << 30);  // addr[56:32] | type=2
    i32x8 g1;
    // data_size=1 (2B) | pad_enable | pad_interval=3 (16 DW) | pad_amount=3 (4 DW)
    g1[0] = (1 << 16) | (1 << 20) | (3 << 22) | (3 << 25);
    g1[1] = (int)(((unsigned)CB & 0xFFFFu) << 16);     // tensor_dim0 = 1024 (low16)
    g1[2] = (int)(((unsigned)(4 * TB * CB / CB) & 0xFFFFu) << 16); // tensor_dim1 = 8192 rows (low16)
    g1[3] = (int)(32u << 16);                          // tile_dim0 = 32 elems
    g1[4] = 128;                                       // tile_dim1 = 128 rows, tile_dim2 = 0
    g1[5] = CB;                                        // tensor_dim0_stride = 1024 (low32)
    g1[6] = 0;                                         // stride high / dim1_stride low
    g1[7] = 0;
    i32x4 z4; z4[0] = 0; z4[1] = 0; z4[2] = 0; z4[3] = 0;
#if defined(__clang_major__) && __clang_major__ >= 23
    i32x8 z8;
#pragma unroll
    for (int j = 0; j < 8; ++j) z8[j] = 0;
    __builtin_amdgcn_tensor_load_to_lds(g0, g1, z4, z4, z8, 0);
#else
    __builtin_amdgcn_tensor_load_to_lds(g0, g1, z4, z4, 0);
#endif
}

// 128x128 output tile GEMM: out = A[M,K] * W[N,K]^T, A fp32 or bf16, W fp32, K=1024.
// 256 threads = 8 waves in 2(M)x4(N); each wave 64x32 -> acc[4][2] fragments.
// USE_TDM: A is bf16 and the A-tile is DMA'd to LDS by the Tensor Data Mover.
template <typename TA, bool USE_TDM>
__device__ __forceinline__ void gemm128x128(const TA* __restrict__ A,
                                            const float* __restrict__ W,
                                            int mBlk, int nBlk, int Kdim,
                                            __bf16* As, __bf16* Bs, v8f acc[4][2]) {
    const int tid  = threadIdx.x;
    const int lane = tid & 31;
    const int wave = tid >> 5;
    const int waveM = wave & 1;
    const int waveN = wave >> 1;

#pragma unroll
    for (int mf = 0; mf < 4; ++mf)
#pragma unroll
        for (int nf = 0; nf < 2; ++nf)
#pragma unroll
            for (int j = 0; j < 8; ++j) acc[mf][nf][j] = 0.0f;

    const unsigned asLds = (unsigned)(size_t)(void*)As;

    for (int k0 = 0; k0 < Kdim; k0 += 32) {
        const int rbase = tid >> 3;         // 0..31
        const int cg    = (tid & 7) * 4;    // 0,4,..28

        if constexpr (USE_TDM) {
            if (wave == 0)
                tdm_load_tile_bf16(reinterpret_cast<const __bf16*>(A) +
                                       (size_t)(mBlk * 128) * Kdim + k0,
                                   asLds);
        } else {
            // Cooperative A fill: 128 rows x 32 cols, fp32 -> bf16
#pragma unroll
            for (int it = 0; it < 4; ++it) {
                int row = rbase + it * 32;
                const TA* srcA = A + (size_t)(mBlk * 128 + row) * Kdim + k0 + cg;
                __bf16* dA = As + row * 40 + cg;
                float4 v = *reinterpret_cast<const float4*>(srcA);
                dA[0] = f2bf(v.x); dA[1] = f2bf(v.y); dA[2] = f2bf(v.z); dA[3] = f2bf(v.w);
            }
        }
        // Cooperative W fill (always fp32 -> bf16), overlaps the TDM transfer
#pragma unroll
        for (int it = 0; it < 4; ++it) {
            int row = rbase + it * 32;
            const float* srcW = W + (size_t)(nBlk * 128 + row) * Kdim + k0 + cg;
            float4 wv = *reinterpret_cast<const float4*>(srcW);
            __bf16* dW = Bs + row * 40 + cg;
            dW[0] = f2bf(wv.x); dW[1] = f2bf(wv.y); dW[2] = f2bf(wv.z); dW[3] = f2bf(wv.w);
        }
        if constexpr (USE_TDM) {
            if (wave == 0) __builtin_amdgcn_s_wait_tensorcnt(0);
        }
        __syncthreads();

        v16bf af[4], bfr[2];
#pragma unroll
        for (int mf = 0; mf < 4; ++mf)
            af[mf] = load_afrag(As + (waveM * 64 + mf * 16) * 40, 40, lane);
#pragma unroll
        for (int nf = 0; nf < 2; ++nf)
            bfr[nf] = load_bfrag(Bs + (waveN * 32 + nf * 16) * 40, 40, lane);
#pragma unroll
        for (int mf = 0; mf < 4; ++mf)
#pragma unroll
            for (int nf = 0; nf < 2; ++nf)
                acc[mf][nf] = wmma_bf16(af[mf], bfr[nf], acc[mf][nf]);
        __syncthreads();
    }
}

// ---------------- Kernel 1: fused QKV projection ----------------
// grid: (M/128=64, N/128=8, 3). z: 0->Q[B,H,T,D], 1->K[B,H,T,D], 2->V[B,H,D,T] (all bf16)
__global__ __launch_bounds__(256) void qkv_kernel(
    const float* __restrict__ x,
    const float* __restrict__ Wq, const float* __restrict__ bq,
    const float* __restrict__ Wk, const float* __restrict__ bk,
    const float* __restrict__ Wv, const float* __restrict__ bv,
    __bf16* __restrict__ Q, __bf16* __restrict__ K, __bf16* __restrict__ V) {
    __shared__ __bf16 As[128 * 40];
    __shared__ __bf16 Bs[128 * 40];
    const int mBlk = blockIdx.x, nBlk = blockIdx.y, z = blockIdx.z;
    const float* W    = (z == 0) ? Wq : (z == 1) ? Wk : Wv;
    const float* bias = (z == 0) ? bq : (z == 1) ? bk : bv;

    v8f acc[4][2];
    gemm128x128<float, false>(x, W, mBlk, nBlk, CB, As, Bs, acc);

    const int tid = threadIdx.x, lane = tid & 31, wave = tid >> 5;
    const int waveM = wave & 1, waveN = wave >> 1;
    const int mhalf = (lane < 16) ? 0 : 8;
    const int n = lane & 15;
#pragma unroll
    for (int mf = 0; mf < 4; ++mf)
#pragma unroll
        for (int nf = 0; nf < 2; ++nf)
#pragma unroll
            for (int r = 0; r < 8; ++r) {
                int row = mBlk * 128 + waveM * 64 + mf * 16 + mhalf + r; // 0..8191
                int col = nBlk * 128 + waveN * 32 + nf * 16 + n;        // 0..1023
                float val = acc[mf][nf][r] + bias[col];
                int b = row >> 11, t = row & (TB - 1);
                int h = col >> 6,  d = col & (DB - 1);
                __bf16 bvv = f2bf(val);
                size_t head = (size_t)(b * 16 + h);
                if (z == 0)      Q[((head * TB + t) << 6) + d] = bvv;
                else if (z == 1) K[((head * TB + t) << 6) + d] = bvv;
                else             V[(head * DB + d) * TB + t]   = bvv;   // transposed
            }
}

// ---------------- Kernel 2: causal flash attention ----------------
// grid: (T/128=16, B*H=64), 256 threads = 8 waves, wave owns 16 query rows.
// K/V fragments streamed straight from global (whole bf16 K/V = 32MB, L2-resident on 192MB L2).
__global__ __launch_bounds__(256) void attn_kernel(
    const __bf16* __restrict__ Q, const __bf16* __restrict__ K,
    const __bf16* __restrict__ V, __bf16* __restrict__ Y) {
    __shared__ __bf16 Plds[8 * 16 * 64];

    const int tid = threadIdx.x, lane = tid & 31, wave = tid >> 5;
    const int bh = blockIdx.y;
    const int b = bh >> 4, h = bh & 15;
    const int qRow0 = blockIdx.x * 128 + wave * 16;

    const size_t head = (size_t)bh * TB * DB;
    const __bf16* Qh = Q + head;
    const __bf16* Kh = K + head;
    const __bf16* Vh = V + head;     // [D,T] within head

    v16bf qa[2];
    qa[0] = load_afrag(Qh + (size_t)qRow0 * DB,      DB, lane);  // d 0..31
    qa[1] = load_afrag(Qh + (size_t)qRow0 * DB + 32, DB, lane);  // d 32..63

    float m[8], l[8];
    v8f o[4];
#pragma unroll
    for (int r = 0; r < 8; ++r) { m[r] = -3.0e38f; l[r] = 0.0f; }
#pragma unroll
    for (int df = 0; df < 4; ++df)
#pragma unroll
        for (int r = 0; r < 8; ++r) o[df][r] = 0.0f;

    __bf16* pw = Plds + wave * 16 * 64;
    const float sc = 0.125f * 1.44269504088896f;   // 1/sqrt(64) * log2(e)
    const int nCh = qRow0 / 64 + 1;

    for (int ch = 0; ch < nCh; ++ch) {
        const int kB = ch * 64;
        // S = Q * K^T for 64 keys (4 fragments of 16 cols, K-dim D=64 -> 2 WMMAs each)
        v8f s[4];
#pragma unroll
        for (int nf = 0; nf < 4; ++nf) {
            v16bf kb0 = load_bfrag(Kh + (size_t)(kB + nf * 16) * DB,      DB, lane);
            v16bf kb1 = load_bfrag(Kh + (size_t)(kB + nf * 16) * DB + 32, DB, lane);
            v8f z;
#pragma unroll
            for (int j = 0; j < 8; ++j) z[j] = 0.0f;
            z = wmma_bf16(qa[0], kb0, z);
            z = wmma_bf16(qa[1], kb1, z);
            s[nf] = z;
        }
        // scale + causal mask (log2 domain)
#pragma unroll
        for (int nf = 0; nf < 4; ++nf)
#pragma unroll
            for (int r = 0; r < 8; ++r) {
                int row = qRow0 + ((lane < 16) ? r : 8 + r);
                int col = kB + nf * 16 + (lane & 15);
                float v = s[nf][r] * sc;
                s[nf][r] = (col <= row) ? v : -3.0e38f;
            }
        // row max across the 16 lanes of each half-wave
        float mloc[8], corr[8], rs[8];
#pragma unroll
        for (int r = 0; r < 8; ++r) {
            float mm = s[0][r];
#pragma unroll
            for (int nf = 1; nf < 4; ++nf) mm = fmaxf(mm, s[nf][r]);
#pragma unroll
            for (int off = 1; off < 16; off <<= 1)
                mm = fmaxf(mm, __shfl_xor(mm, off, 32));
            mloc[r] = mm;
        }
#pragma unroll
        for (int r = 0; r < 8; ++r) {
            float mn = fmaxf(m[r], mloc[r]);
            corr[r] = exp2f(m[r] - mn);
            m[r] = mn;
        }
        // P = exp2(S - m), row sums
#pragma unroll
        for (int r = 0; r < 8; ++r) {
            float a = 0.0f;
#pragma unroll
            for (int nf = 0; nf < 4; ++nf) {
                float p = exp2f(s[nf][r] - m[r]);
                s[nf][r] = p;
                a += p;
            }
#pragma unroll
            for (int off = 1; off < 16; off <<= 1) a += __shfl_xor(a, off, 32);
            rs[r] = a;
        }
#pragma unroll
        for (int r = 0; r < 8; ++r) l[r] = l[r] * corr[r] + rs[r];
#pragma unroll
        for (int df = 0; df < 4; ++df)
#pragma unroll
            for (int r = 0; r < 8; ++r) o[df][r] *= corr[r];

        // C-layout -> A-layout for P via per-wave LDS slab (same-wave DS ops are in-order)
#pragma unroll
        for (int nf = 0; nf < 4; ++nf)
#pragma unroll
            for (int r = 0; r < 8; ++r)
                pw[((lane < 16) ? r : 8 + r) * 64 + nf * 16 + (lane & 15)] = f2bf(s[nf][r]);
        asm volatile("s_wait_dscnt 0" ::: "memory");

        v16bf pa0 = load_afrag(pw,      64, lane);   // keys kB+0..31
        v16bf pa1 = load_afrag(pw + 32, 64, lane);   // keys kB+32..63

        // O += P * V  (V^T layout: lane's 16 K-values are two 8-element runs along T (k 8h.., 16+8h..))
#pragma unroll
        for (int df = 0; df < 4; ++df) {
            v16bf vb0 = load_bfrag(Vh + (size_t)(df * 16) * TB + kB,      TB, lane);
            v16bf vb1 = load_bfrag(Vh + (size_t)(df * 16) * TB + kB + 32, TB, lane);
            o[df] = wmma_bf16(pa0, vb0, o[df]);
            o[df] = wmma_bf16(pa1, vb1, o[df]);
        }
        asm volatile("" ::: "memory");
    }

    // epilogue: normalize, write Y as bf16 [B*T, C] (row-major, ready as GEMM-A)
#pragma unroll
    for (int df = 0; df < 4; ++df)
#pragma unroll
        for (int r = 0; r < 8; ++r) {
            int t = qRow0 + ((lane < 16) ? r : 8 + r);
            int col = h * DB + df * 16 + (lane & 15);
            float y = o[df][r] * (1.0f / l[r]);
            Y[((size_t)b * TB + t) * CB + col] = f2bf(y);
        }
}

// ---------------- Kernel 3: output projection (bf16 A via TDM, fp32 out) ----------------
__global__ __launch_bounds__(256) void proj_kernel(
    const __bf16* __restrict__ Y, const float* __restrict__ Wp,
    const float* __restrict__ bp, float* __restrict__ out) {
    __shared__ __bf16 As[128 * 40];
    __shared__ __bf16 Bs[128 * 40];
    v8f acc[4][2];
    gemm128x128<__bf16, true>(Y, Wp, blockIdx.x, blockIdx.y, CB, As, Bs, acc);

    const int tid = threadIdx.x, lane = tid & 31, wave = tid >> 5;
    const int waveM = wave & 1, waveN = wave >> 1;
    const int mhalf = (lane < 16) ? 0 : 8;
    const int n = lane & 15;
#pragma unroll
    for (int mf = 0; mf < 4; ++mf)
#pragma unroll
        for (int nf = 0; nf < 2; ++nf)
#pragma unroll
            for (int r = 0; r < 8; ++r) {
                int row = blockIdx.x * 128 + waveM * 64 + mf * 16 + mhalf + r;
                int col = blockIdx.y * 128 + waveN * 32 + nf * 16 + n;
                out[(size_t)row * CB + col] = acc[mf][nf][r] + bp[col];
            }
}

extern "C" void kernel_launch(void* const* d_in, const int* in_sizes, int n_in,
                              void* d_out, int out_size, void* d_ws, size_t ws_size,
                              hipStream_t stream) {
    (void)in_sizes; (void)n_in; (void)out_size; (void)ws_size;
    const float* x  = (const float*)d_in[0];
    const float* Wq = (const float*)d_in[1];
    const float* bq = (const float*)d_in[2];
    const float* Wk = (const float*)d_in[3];
    const float* bk = (const float*)d_in[4];
    const float* Wv = (const float*)d_in[5];
    const float* bv = (const float*)d_in[6];
    const float* Wp = (const float*)d_in[7];
    const float* bp = (const float*)d_in[8];
    float* out = (float*)d_out;

    char* ws = (char*)d_ws;
    const size_t elems = (size_t)4 * TB * CB;          // 8,388,608
    __bf16* Qb = (__bf16*)(ws);
    __bf16* Kb = (__bf16*)(ws + elems * 2);
    __bf16* Vb = (__bf16*)(ws + elems * 4);
    __bf16* Yb = (__bf16*)(ws + elems * 6);            // 64 MB total workspace

    qkv_kernel<<<dim3(64, 8, 3), 256, 0, stream>>>(x, Wq, bq, Wk, bk, Wv, bv, Qb, Kb, Vb);
    attn_kernel<<<dim3(16, 64), 256, 0, stream>>>(Qb, Kb, Vb, Yb);
    proj_kernel<<<dim3(64, 8), 256, 0, stream>>>(Yb, Wp, bp, out);
}
